// TomatoModel_1425929142386
// MI455X (gfx1250) — matrix operand order repaired and checked by reading, not compiled, not run on hardware
//
#include <hip/hip_runtime.h>
#include <hip/hip_bf16.h>
#include <math.h>

typedef _Float16 h16;
typedef __attribute__((ext_vector_type(16))) _Float16 v16h;
typedef __attribute__((ext_vector_type(8)))  _Float16 v8h;
typedef __attribute__((ext_vector_type(8)))  float    v8f;
typedef __attribute__((ext_vector_type(4)))  unsigned int u32x4;
typedef __attribute__((ext_vector_type(8)))  int i32x8;
typedef __attribute__((ext_vector_type(4)))  int i32x4;

#define BB    64
#define TT    48
#define HID   512
#define COMBP 576   // 560 padded to 576 (18 k-tiles of 32)

// ---------------------------------------------------------------------------
// WMMA helper: D = A(16x32 f16) * B(32x16 f16) + C(16x16 f32)
// ---------------------------------------------------------------------------
__device__ __forceinline__ v8f wmma16(v16h a, v16h b, v8f c) {
  return __builtin_amdgcn_wmma_f32_16x16x32_f16(
      false, a, false, b, (short)0, c, false, false);
}

// A fragment, row-major [M x K], leading dim ld (halves). Contiguous k = 16h + s layout.
__device__ __forceinline__ v16h frag_a(const h16* base, int ld, int m0, int k0, int lane) {
  int m  = m0 + (lane & 15);
  int kh = (lane < 16) ? 0 : 16;
  const h16* p = base + (size_t)m * ld + k0 + kh;
  v8h lo = *(const v8h*)(p);
  v8h hi = *(const v8h*)(p + 8);
  v16h r;
#pragma unroll
  for (int i = 0; i < 8; i++) { r[i] = lo[i]; r[8 + i] = hi[i]; }
  return r;
}

// B fragment from an [N x K] row-major matrix (computes X*W^T).
__device__ __forceinline__ v16h frag_b(const h16* base, int ld, int n0, int k0, int lane) {
  int n  = n0 + (lane & 15);
  int ko = (lane < 16) ? 0 : 16;
  const h16* p = base + (size_t)n * ld + k0 + ko;
  v8h lo = *(const v8h*)(p);
  v8h hi = *(const v8h*)(p + 8);
  v16h r;
#pragma unroll
  for (int i = 0; i < 8; i++) { r[i] = lo[i]; r[8 + i] = hi[i]; }
  return r;
}

// ---------------------------------------------------------------------------
// Tensor Data Mover: 1-D (1 x nelem) 2-byte-element tile, global -> LDS.
// D# packed per ISA ch.8 (group0: count/lds_addr/global_addr/type,
// group1: data_size=2B, tensor_dim0=nelem, tensor_dim1=1, tile=nelem x 1).
// This toolchain exposes the 6-arg builtin (g0, g1, g2, g3, g4, cpol).
// ---------------------------------------------------------------------------
__device__ __forceinline__ void tdm_load_1d(unsigned int lds_off, const void* gptr,
                                            unsigned int nelem) {
  unsigned long long ga = (unsigned long long)(uintptr_t)gptr;
  u32x4 g0;
  g0[0] = 1u;                                        // count=1, user desc
  g0[1] = lds_off;                                   // lds_addr
  g0[2] = (unsigned int)ga;                          // global_addr[31:0]
  g0[3] = (unsigned int)(ga >> 32) | (2u << 30);     // global_addr[56:32] | type=2
  i32x8 g1;
  g1[0] = (int)(1u << 16);                           // data_size=1 (2 bytes)
  g1[1] = (int)((nelem & 0xFFFFu) << 16);            // tensor_dim0[15:0] @ bits63:48
  g1[2] = (int)((nelem >> 16) & 0xFFFFu) | (int)(1u << 16); // dim0 hi | tensor_dim1=1
  g1[3] = (int)((nelem & 0xFFFFu) << 16);            // tile_dim0 @ bits127:112
  g1[4] = 1;                                         // tile_dim1 = 1
  g1[5] = (int)nelem;                                // tensor_dim0_stride lo32
  g1[6] = 0;
  g1[7] = 0;
  i32x4 z4 = {0, 0, 0, 0};
  i32x8 z8 = {0, 0, 0, 0, 0, 0, 0, 0};
  __builtin_amdgcn_tensor_load_to_lds(g0, g1, z4, z4, z8, 0);
}

// ---------------------------------------------------------------------------
// Weight conversion kernels
// ---------------------------------------------------------------------------
__global__ void cvt_pad_kernel(const float* __restrict__ src, h16* __restrict__ dst,
                               int rows, int ksrc, int kdst) {
  int idx = blockIdx.x * 256 + threadIdx.x;
  if (idx >= rows * kdst) return;
  int r = idx / kdst, k = idx - r * kdst;
  dst[idx] = (h16)((k < ksrc) ? src[r * ksrc + k] : 0.0f);
}

// repack conv weights with per-channel K padding:
// khat = ic*kpd + rem ; value = rem < kps ? src[r*ksrc + ic*kps + rem] : 0
__global__ void cvt_conv_kernel(const float* __restrict__ src, h16* __restrict__ dst,
                                int rows, int ksrc, int kps, int kpd, int kdst) {
  int idx = blockIdx.x * 256 + threadIdx.x;
  if (idx >= rows * kdst) return;
  int r = idx / kdst, khat = idx - r * kdst;
  int ic = khat / kpd, rem = khat - ic * kpd;
  float v = 0.0f;
  if (rem < kps) v = src[r * ksrc + ic * kps + rem];
  dst[idx] = (h16)v;
}

__global__ void pack_bias_kernel(const float* __restrict__ bi, const float* __restrict__ bf,
                                 const float* __restrict__ bc, const float* __restrict__ bo,
                                 float* __restrict__ bg) {
  int idx = blockIdx.x * 256 + threadIdx.x;
  if (idx >= 4 * HID) return;
  int g = idx >> 9, n = idx & 511;
  const float* p = (g == 0) ? bi : (g == 1) ? bf : (g == 2) ? bc : bo;
  bg[idx] = p[n];
}

__global__ void datas_cvt_kernel(const float* __restrict__ datas, h16* __restrict__ xseq) {
  int idx = blockIdx.x * 256 + threadIdx.x;   // B*T*32
  if (idx >= BB * TT * 32) return;
  int b = idx / (TT * 32);
  int rem = idx - b * TT * 32;
  int t = rem / 32, k = rem - t * 32;
  xseq[((size_t)t * BB + b) * 48 + 16 + k] = (h16)datas[idx];
}

// ---------------------------------------------------------------------------
// conv1 (16oc, 3x11x11 s4) + ReLU + maxpool 3x3 s2, fused.
// grid (3072 images, 4 pool-row bands). Implicit GEMM M=16, K=384 (khat
// layout ic*128+r, zero-padded weights annihilate pad lanes -> no branches).
// ---------------------------------------------------------------------------
#define S_IMG_SZ (3 * 43 * 116 + 128)
__global__ __launch_bounds__(256) void conv1_pool_kernel(
    const float* __restrict__ img, const h16* __restrict__ w1,
    const float* __restrict__ b1, h16* __restrict__ out1) {
  __shared__ h16   s_img[S_IMG_SZ];
  __shared__ h16   s_w[16 * 384];
  __shared__ float s_out[16 * 244];

  const int image = blockIdx.x;
  const int band  = blockIdx.y;
  const int pr0_t[4] = {0, 4, 7, 10};
  const int pr1_t[4] = {3, 6, 9, 12};
  int pr0 = pr0_t[band], pr1 = pr1_t[band];
  int c0 = 2 * pr0, c1 = 2 * pr1 + 2;
  int nCr = c1 - c0 + 1;            // conv rows in band (<= 9)
  int i0  = 4 * c0;                 // first input row
  int nIr = 4 * (nCr - 1) + 11;     // input rows (<= 43)
  int tid = threadIdx.x;

  // zero-fill (pad reads must hit finite zeros: WMMA propagates NaN)
  for (int idx = tid; idx < S_IMG_SZ; idx += 256) s_img[idx] = (h16)0.0f;
  __syncthreads();

  // stage input band f32 -> f16
  const float* ibase = img + (size_t)image * 3 * 115 * 115;
  int total = 3 * nIr * 115;
  for (int idx = tid; idx < total; idx += 256) {
    int ch  = idx / (nIr * 115);
    int rem = idx - ch * nIr * 115;
    int r   = rem / 115;
    int ccl = rem - r * 115;
    s_img[(ch * 43 + r) * 116 + ccl] = (h16)ibase[(ch * 115 + (i0 + r)) * 115 + ccl];
  }
  for (int idx = tid; idx < 16 * 384; idx += 256) s_w[idx] = w1[idx];
  __syncthreads();

  int nPos   = nCr * 27;
  int nTiles = (nPos + 15) / 16;
  int wave = tid >> 5, lane = tid & 31;

  for (int tile = wave; tile < nTiles; tile += 8) {
    int posBase = tile * 16;
    int pos = posBase + (lane & 15);
    if (pos >= nPos) pos = nPos - 1;   // clamp; writes guarded below
    int cr = pos / 27;
    int pw = pos - cr * 27;
    int base = (4 * cr) * 116 + 4 * pw;   // per-lane constant
    v8f c;
#pragma unroll
    for (int j = 0; j < 8; j++) c[j] = b1[j + ((lane < 16) ? 0 : 8)];

    for (int kt = 0; kt < 12; kt++) {
      int k0 = kt * 32;
      v16h a = frag_a(s_w, 384, 0, k0, lane);
      int kb = k0 + ((lane < 16) ? 0 : 16);
      int ic = kb >> 7;                   // 16-run never crosses a channel
      int remb = kb & 127;
      const h16* sp = s_img + ic * (43 * 116) + base;
      v16h b;
#pragma unroll
      for (int j = 0; j < 16; j++) {
        int rem = remb + j;               // <= 127
        int kr = rem / 11;                // pad rows (kr==11) have zero weights
        int kc = rem - kr * 11;
        b[j] = sp[kr * 116 + kc];
      }
      c = wmma16(a, b, c);
    }
    if (posBase + (lane & 15) < nPos) {
#pragma unroll
      for (int j = 0; j < 8; j++) {
        int m = j + ((lane < 16) ? 0 : 8);
        float v = c[j];
        s_out[m * 244 + pos] = (v > 0.0f) ? v : 0.0f;
      }
    }
  }
  __syncthreads();

  // maxpool 3x3 s2 -> f16
  int nPr = pr1 - pr0 + 1;
  int tot2 = 16 * nPr * 13;
  for (int idx = tid; idx < tot2; idx += 256) {
    int oc  = idx / (nPr * 13);
    int rem = idx - oc * nPr * 13;
    int prl = rem / 13;
    int pw  = rem - prl * 13;
    int pr  = pr0 + prl;
    float mx = -1e30f;
#pragma unroll
    for (int dy = 0; dy < 3; dy++)
#pragma unroll
      for (int dx = 0; dx < 3; dx++) {
        int crl = 2 * pr + dy - c0;
        float v = s_out[oc * 244 + crl * 27 + 2 * pw + dx];
        if (v > mx) mx = v;
      }
    out1[((size_t)image * 16 + oc) * 169 + pr * 13 + pw] = (h16)mx;
  }
}

// ---------------------------------------------------------------------------
// conv2 (32oc, 16x3x3 pad1) + ReLU + maxpool, one WG per image.
// Activations staged with zero halo [16][16][15]; weights staged via TDM.
// khat layout ic*10+r (pad lane zero-weighted) -> branch-free gather.
// ---------------------------------------------------------------------------
__global__ __launch_bounds__(256) void conv2_pool_kernel(
    const h16* __restrict__ in1, const h16* __restrict__ w2,
    const float* __restrict__ b2, h16* __restrict__ out2) {
  __shared__ h16   s_in[16 * 16 * 15];   // halo-padded 13x13 per channel
  __shared__ h16   s_w[32 * 160];
  __shared__ float s_out[32 * 172];

  int image = blockIdx.x, tid = threadIdx.x;

  // TDM: async DMA of the weight tile into LDS (one issue, wave 0)
  if (tid < 32) {
    tdm_load_1d((unsigned int)(uintptr_t)&s_w[0], w2, 32u * 160u);
  }

  for (int idx = tid; idx < 16 * 16 * 15; idx += 256) s_in[idx] = (h16)0.0f;
  __syncthreads();

  const h16* ib = in1 + (size_t)image * 16 * 169;
  for (int idx = tid; idx < 16 * 169; idx += 256) {
    int ic = idx / 169;
    int rem = idx - ic * 169;
    int y = rem / 13, x = rem - y * 13;
    s_in[ic * 240 + (y + 1) * 15 + (x + 1)] = ib[idx];
  }
  if (tid < 32) __builtin_amdgcn_s_wait_tensorcnt(0);
  __syncthreads();

  int wave = tid >> 5, lane = tid & 31;
  for (int t = wave; t < 22; t += 8) {          // 2 M-tiles x 11 N-tiles
    int mt = t & 1, nt = t >> 1;
    int m0 = mt * 16;
    int pos = nt * 16 + (lane & 15);
    if (pos > 168) pos = 168;
    int py = pos / 13, px = pos - py * 13;
    int base = py * 15 + px;                    // halo origin (y+kr)(x+kc)
    v8f c;
#pragma unroll
    for (int j = 0; j < 8; j++) c[j] = b2[m0 + j + ((lane < 16) ? 0 : 8)];

    for (int kt = 0; kt < 5; kt++) {
      int k0 = kt * 32;
      v16h a = frag_a(s_w, 160, m0, k0, lane);
      int kb = k0 + ((lane < 16) ? 0 : 16);
      v16h b;
#pragma unroll
      for (int j = 0; j < 16; j++) {
        int k = kb + j;
        int ic = k / 10;
        int rem = k - ic * 10;                  // rem==9 is pad (zero weight)
        int kr = rem / 3;
        int kc = rem - kr * 3;
        b[j] = s_in[ic * 240 + base + kr * 15 + kc];
      }
      c = wmma16(a, b, c);
    }
    if (nt * 16 + (lane & 15) < 169) {
#pragma unroll
      for (int j = 0; j < 8; j++) {
        int m = m0 + j + ((lane < 16) ? 0 : 8);
        float v = c[j];
        s_out[m * 172 + pos] = (v > 0.0f) ? v : 0.0f;
      }
    }
  }
  __syncthreads();

  for (int idx = tid; idx < 32 * 36; idx += 256) {
    int oc  = idx / 36;
    int rem = idx - oc * 36;
    int py = rem / 6, px = rem - py * 6;
    float mx = -1e30f;
#pragma unroll
    for (int dy = 0; dy < 3; dy++)
#pragma unroll
      for (int dx = 0; dx < 3; dx++) {
        float v = s_out[oc * 172 + (2 * py + dy) * 13 + 2 * px + dx];
        if (v > mx) mx = v;
      }
    out2[(size_t)image * 1152 + idx] = (h16)mx;
  }
}

// ---------------------------------------------------------------------------
// pl: feats = relu(X[3072x1152] @ plw^T[16x1152]) -> xseq[t][b][0:16]
// ---------------------------------------------------------------------------
__global__ __launch_bounds__(32) void pl_kernel(const h16* __restrict__ x,
                                                const h16* __restrict__ w,
                                                const float* __restrict__ bias,
                                                h16* __restrict__ xseq) {
  int r0 = blockIdx.x * 16;
  int lane = threadIdx.x & 31;
  int n = lane & 15;
  v8f c;
  float bn = bias[n];
#pragma unroll
  for (int j = 0; j < 8; j++) c[j] = bn;
  for (int kt = 0; kt < 36; kt++) {
    v16h a = frag_a(x, 1152, r0, kt * 32, lane);
    v16h b = frag_b(w, 1152, 0, kt * 32, lane);
    c = wmma16(a, b, c);
  }
#pragma unroll
  for (int j = 0; j < 8; j++) {
    int r = r0 + j + ((lane < 16) ? 0 : 8);
    int bidx = r / TT, t = r - bidx * TT;
    float v = c[j];
    xseq[((size_t)t * BB + bidx) * 48 + n] = (h16)((v > 0.0f) ? v : 0.0f);
  }
}

// ---------------------------------------------------------------------------
// LSTM init: comb = [x_0 | 0 | pad], h = c = 0
// ---------------------------------------------------------------------------
__global__ void lstm_init_kernel(const h16* __restrict__ xseq, h16* __restrict__ comb,
                                 float* __restrict__ hbuf, float* __restrict__ cbuf) {
  int idx = blockIdx.x * 256 + threadIdx.x;   // 64*576
  if (idx < BB * COMBP) {
    int b = idx / COMBP, k = idx - b * COMBP;
    h16 v = (h16)0.0f;
    if (k < 48) v = xseq[(size_t)b * 48 + k];   // t = 0
    comb[idx] = v;
  }
  if (idx < BB * HID) { hbuf[idx] = 0.0f; cbuf[idx] = 0.0f; }
}

// ---------------------------------------------------------------------------
// Gates GEMM: gatebuf[n][m] = bias[n] + comb[64x576] @ Wg^T[2048x576]
// grid 16, block 256 (8 waves x 4 tiles). Weights L2-resident.
// ---------------------------------------------------------------------------
__global__ __launch_bounds__(256) void gates_kernel(const h16* __restrict__ comb,
                                                    const h16* __restrict__ Wg,
                                                    const float* __restrict__ bg,
                                                    float* __restrict__ gatebuf) {
  int wgi = blockIdx.x;
  int tid = threadIdx.x;
  int wave = tid >> 5, lane = tid & 31;
#pragma unroll
  for (int i = 0; i < 4; i++) {
    int tileId = wave * 4 + i;
    int m0 = (tileId & 3) * 16;
    int n0 = wgi * 128 + (tileId >> 2) * 16;
    int n  = n0 + (lane & 15);
    v8f c;
    float bn = bg[n];
#pragma unroll
    for (int j = 0; j < 8; j++) c[j] = bn;
    for (int kt = 0; kt < 18; kt++) {
      __builtin_prefetch(Wg + (size_t)n * COMBP + (kt + 1) * 32, 0, 3);
      v16h a = frag_a(comb, COMBP, m0, kt * 32, lane);
      v16h b = frag_b(Wg, COMBP, n0, kt * 32, lane);
      c = wmma16(a, b, c);
    }
    float* p = gatebuf + (size_t)n * 64 + m0 + ((lane < 16) ? 0 : 8);
    *(float4*)(p)     = make_float4(c[0], c[1], c[2], c[3]);
    *(float4*)(p + 4) = make_float4(c[4], c[5], c[6], c[7]);
  }
}

// ---------------------------------------------------------------------------
// LSTM cell elementwise + masking; also stages comb for step t+1.
// ---------------------------------------------------------------------------
__global__ __launch_bounds__(256) void lstm_cell_kernel(
    const float* __restrict__ gatebuf, const long long* __restrict__ lens,
    const h16* __restrict__ xseq, int t,
    float* __restrict__ hbuf, float* __restrict__ cbuf, h16* __restrict__ comb) {
  int idx = blockIdx.x * 256 + threadIdx.x;
  if (idx >= BB * HID) return;
  int b = idx & 63;
  int n = idx >> 6;
  float gi = gatebuf[(size_t)(0 * HID + n) * 64 + b];
  float gf = gatebuf[(size_t)(1 * HID + n) * 64 + b];
  float gc = gatebuf[(size_t)(2 * HID + n) * 64 + b];
  float go = gatebuf[(size_t)(3 * HID + n) * 64 + b];
  float i_t = 1.0f / (1.0f + expf(-gi));
  float f_t = 1.0f / (1.0f + expf(-gf));
  float ch  = tanhf(gc);
  float o_t = 1.0f / (1.0f + expf(-go));
  int hidx = b * HID + n;
  float c_old = cbuf[hidx], h_old = hbuf[hidx];
  float c_new = f_t * c_old + i_t * ch;
  float h_new = o_t * tanhf(c_new);
  bool m = ((long long)t) < lens[b];
  float hN = m ? h_new : h_old;
  float cN = m ? c_new : c_old;
  hbuf[hidx] = hN;
  cbuf[hidx] = cN;
  comb[(size_t)b * COMBP + 48 + n] = (h16)hN;
  if (n < 48 && (t + 1) < TT)
    comb[(size_t)b * COMBP + n] = xseq[((size_t)(t + 1) * BB + b) * 48 + n];
}

// ---------------------------------------------------------------------------
// classifier + output: out[0:256] = y(64x4), out[256:] = h(64x512)
// ---------------------------------------------------------------------------
__global__ __launch_bounds__(256) void cls_kernel(const float* __restrict__ hbuf,
                                                  const float* __restrict__ cw,
                                                  const float* __restrict__ cb,
                                                  float* __restrict__ out) {
  int tid = threadIdx.x;
  if (blockIdx.x == 0) {
    int b = tid >> 2, o = tid & 3;
    float s = cb[o];
    for (int n = 0; n < HID; n++) s += hbuf[b * HID + n] * cw[o * HID + n];
    out[b * 4 + o] = s;
  } else {
    int idx = (blockIdx.x - 1) * 256 + tid;
    if (idx < BB * HID) out[256 + idx] = hbuf[idx];
  }
}

// ---------------------------------------------------------------------------
// Host launcher
// ---------------------------------------------------------------------------
extern "C" void kernel_launch(void* const* d_in, const int* in_sizes, int n_in,
                              void* d_out, int out_size, void* d_ws, size_t ws_size,
                              hipStream_t stream) {
  const float*      datas   = (const float*)d_in[0];
  const float*      img     = (const float*)d_in[1];
  const long long*  lens    = (const long long*)d_in[2];
  const float*      conv1_w = (const float*)d_in[3];
  const float*      conv1_b = (const float*)d_in[4];
  const float*      conv2_w = (const float*)d_in[5];
  const float*      conv2_b = (const float*)d_in[6];
  const float*      pl_w    = (const float*)d_in[7];
  const float*      pl_b    = (const float*)d_in[8];
  const float*      ig_w    = (const float*)d_in[9];
  const float*      ig_b    = (const float*)d_in[10];
  const float*      fg_w    = (const float*)d_in[11];
  const float*      fg_b    = (const float*)d_in[12];
  const float*      cg_w    = (const float*)d_in[13];
  const float*      cg_b    = (const float*)d_in[14];
  const float*      og_w    = (const float*)d_in[15];
  const float*      og_b    = (const float*)d_in[16];
  const float*      cls_w   = (const float*)d_in[17];
  const float*      cls_b   = (const float*)d_in[18];
  float* out = (float*)d_out;

  char* ws = (char*)d_ws;
  size_t off = 0;
  auto take = [&](size_t bytes) {
    size_t o = off;
    off = (off + bytes + 255) & ~(size_t)255;
    return o;
  };
  h16*   w1h  = (h16*)(ws + take((size_t)16 * 384 * 2));
  h16*   w2h  = (h16*)(ws + take((size_t)32 * 160 * 2));
  h16*   plwh = (h16*)(ws + take((size_t)16 * 1152 * 2));
  h16*   Wg   = (h16*)(ws + take((size_t)4 * HID * COMBP * 2));
  float* bg   = (float*)(ws + take((size_t)4 * HID * 4));
  h16*   out1 = (h16*)(ws + take((size_t)3072 * 16 * 169 * 2));
  h16*   out2 = (h16*)(ws + take((size_t)3072 * 1152 * 2));
  h16*   xseq = (h16*)(ws + take((size_t)TT * BB * 48 * 2));
  h16*   comb = (h16*)(ws + take((size_t)BB * COMBP * 2));
  float* gbuf = (float*)(ws + take((size_t)4 * HID * 64 * 4));
  float* hbuf = (float*)(ws + take((size_t)BB * HID * 4));
  float* cbuf = (float*)(ws + take((size_t)BB * HID * 4));
  (void)ws_size; (void)in_sizes; (void)n_in; (void)out_size;

  // weight conversion (khat-padded conv layouts)
  cvt_conv_kernel<<<(16 * 384 + 255) / 256, 256, 0, stream>>>(conv1_w, w1h, 16, 363, 121, 128, 384);
  cvt_conv_kernel<<<(32 * 160 + 255) / 256, 256, 0, stream>>>(conv2_w, w2h, 32, 144, 9, 10, 160);
  cvt_pad_kernel<<<(16 * 1152 + 255) / 256, 256, 0, stream>>>(pl_w, plwh, 16, 1152, 1152);
  const float* gw[4] = {ig_w, fg_w, cg_w, og_w};
  for (int g = 0; g < 4; g++)
    cvt_pad_kernel<<<(HID * COMBP + 255) / 256, 256, 0, stream>>>(
        gw[g], Wg + (size_t)g * HID * COMBP, HID, 560, COMBP);
  pack_bias_kernel<<<8, 256, 0, stream>>>(ig_b, fg_b, cg_b, og_b, bg);
  datas_cvt_kernel<<<(BB * TT * 32 + 255) / 256, 256, 0, stream>>>(datas, xseq);

  // conv stack
  conv1_pool_kernel<<<dim3(3072, 4), 256, 0, stream>>>(img, w1h, conv1_b, out1);
  conv2_pool_kernel<<<3072, 256, 0, stream>>>(out1, w2h, conv2_b, out2);
  pl_kernel<<<3072 / 16, 32, 0, stream>>>(out2, plwh, pl_b, xseq);

  // LSTM
  lstm_init_kernel<<<(BB * COMBP + 255) / 256, 256, 0, stream>>>(xseq, comb, hbuf, cbuf);
  for (int t = 0; t < TT; t++) {
    gates_kernel<<<16, 256, 0, stream>>>(comb, Wg, bg, gbuf);
    lstm_cell_kernel<<<(BB * HID + 255) / 256, 256, 0, stream>>>(
        gbuf, lens, xseq, t, hbuf, cbuf, comb);
  }

  // classifier + output
  cls_kernel<<<1 + (BB * HID + 255) / 256, 256, 0, stream>>>(hbuf, cls_w, cls_b, out);
}
